// MultiHeadAttention_12000138625748
// MI455X (gfx1250) — compile-verified, matrix operand order checked by reading, not run on hardware
//
#include <hip/hip_runtime.h>
#include <hip/hip_bf16.h>

// ---------------------------------------------------------------------------
// MHA forward for MI455X (gfx1250): bf16 WMMA (f32 accum) + flash-attention,
// TDM (tensor_load_to_lds) double-buffered tile staging, DPP16 row reductions.
// B=4, S=2048, H=1024, NH=16, HD=64
// ---------------------------------------------------------------------------

#define B_  4
#define S_  2048
#define H_  1024
#define NH_ 16
#define HD_ 64
#define M_  (B_ * S_)   // 8192 rows

typedef __attribute__((ext_vector_type(16))) __bf16 v16bf;
typedef __attribute__((ext_vector_type(8)))  __bf16 v8bf;
typedef __attribute__((ext_vector_type(8)))  float  v8f;
typedef __attribute__((ext_vector_type(4)))  unsigned int uint4v;
typedef __attribute__((ext_vector_type(4)))  unsigned int u32x4;
typedef __attribute__((ext_vector_type(8)))  int i32x8;
typedef __attribute__((ext_vector_type(4)))  int i32x4;

#if __has_builtin(__builtin_amdgcn_tensor_load_to_lds) && \
    __has_builtin(__builtin_amdgcn_s_wait_tensorcnt)
#define USE_TDM 1
#else
#define USE_TDM 0
#endif

// ---------------------------------------------------------------------------
// Fragment load: CDNA5 WMMA A/B register layout for 16x32 (bf16) operands.
// lane<16  -> k = 0..7  and 16..23 of its row
// lane>=16 -> k = 8..15 and 24..31 of its row
// rowp: 32 contiguous bf16 (one row, k-major), 16B aligned.
// ---------------------------------------------------------------------------
__device__ __forceinline__ v16bf load_frag16(const __bf16* rowp, int half) {
    union { uint4v u[2]; v16bf v; } t;
    t.u[0] = *(const uint4v*)(rowp + half * 8);
    t.u[1] = *(const uint4v*)(rowp + 16 + half * 8);
    return t.v;
}

// ---------------------------------------------------------------------------
// DPP16 butterfly reductions within each 16-lane half (matches the WMMA C
// fragment row layout: lanes 0-15 = rows 0..7, lanes 16-31 = rows 8..15).
// Pure VALU: no LDS, no dscnt waits (vs ds_bpermute).
// ---------------------------------------------------------------------------
template <int CTRL>
__device__ __forceinline__ float dpp_step(float x) {
    return __int_as_float(__builtin_amdgcn_update_dpp(
        0, __float_as_int(x), CTRL, 0xF, 0xF, true));
}
__device__ __forceinline__ float row16_max(float x) {
    x = fmaxf(x, dpp_step<0x0B1>(x));   // quad_perm [1,0,3,2]
    x = fmaxf(x, dpp_step<0x04E>(x));   // quad_perm [2,3,0,1]
    x = fmaxf(x, dpp_step<0x141>(x));   // row_half_mirror
    x = fmaxf(x, dpp_step<0x140>(x));   // row_mirror
    return x;
}
__device__ __forceinline__ float row16_sum(float x) {
    x += dpp_step<0x0B1>(x);
    x += dpp_step<0x04E>(x);
    x += dpp_step<0x141>(x);
    x += dpp_step<0x140>(x);
    return x;
}

#if USE_TDM
// ---------------------------------------------------------------------------
// Tensor Data Mover: 2D bf16 tile (tile_d0 contiguous elems x tile_d1 rows,
// row stride = stride_elems) from global into LDS at lds_off.
// D# layout per CDNA5 ISA 8.3/8.4: group0 = {flags, lds_addr, global_addr,
// type=2}; group1 = {data_size=2B, tensor dims (large: no OOB), tile dims,
// dim0 stride}. Groups 2/3/extra zero (2D tensor).
// ---------------------------------------------------------------------------
__device__ __forceinline__ unsigned lds_off_of(const void* p) {
    // LDS aperture: flat addr[31:0] is the LDS byte offset.
    return (unsigned)(uintptr_t)p;
}
__device__ __forceinline__ void tdm_load_2d_bf16(unsigned lds_off,
                                                 const __bf16* gptr,
                                                 unsigned tile_d0,
                                                 unsigned tile_d1,
                                                 unsigned stride_elems) {
    unsigned long long ga = (unsigned long long)(uintptr_t)gptr;
    u32x4 g0;
    g0[0] = 1u;                                                // count=1, user D#
    g0[1] = lds_off;                                           // lds_addr
    g0[2] = (unsigned)ga;                                      // global_addr lo
    g0[3] = (unsigned)((ga >> 32) & 0x01FFFFFFu) | (2u << 30); // addr[56:32] | type=2
    i32x8 g1;
    const unsigned td0 = 0x40000000u, td1 = 0x40000000u;       // huge: never OOB
    g1[0] = (int)(1u << 16);                                   // data_size = 2 bytes
    g1[1] = (int)((td0 & 0xFFFFu) << 16);                      // tensor_dim0 lo
    g1[2] = (int)(((td0 >> 16) & 0xFFFFu) | ((td1 & 0xFFFFu) << 16));
    g1[3] = (int)(((td1 >> 16) & 0xFFFFu) | (tile_d0 << 16));  // tile_dim0
    g1[4] = (int)(tile_d1 & 0xFFFFu);                          // tile_dim1 (dim2=0)
    g1[5] = (int)stride_elems;                                 // dim0 stride lo
    g1[6] = 0;                                                 // stride hi, dim1 stride
    g1[7] = 0;
    i32x4 gz4 = {0, 0, 0, 0};
    i32x8 gz8 = {0, 0, 0, 0, 0, 0, 0, 0};
    __builtin_amdgcn_tensor_load_to_lds(g0, g1, gz4, gz4, gz8, 0);
}
#endif

// ---------------------------------------------------------------------------
// fp32 -> bf16 conversion, 8 elements/thread
// ---------------------------------------------------------------------------
__global__ void CvtBf16Kernel(const float* __restrict__ src,
                              __bf16* __restrict__ dst, int n) {
    int i = (blockIdx.x * blockDim.x + threadIdx.x) * 8;
    if (i >= n) return;
    float4 f0 = *(const float4*)(src + i);
    float4 f1 = *(const float4*)(src + i + 4);
    v8bf o;
    o[0] = (__bf16)f0.x; o[1] = (__bf16)f0.y;
    o[2] = (__bf16)f0.z; o[3] = (__bf16)f0.w;
    o[4] = (__bf16)f1.x; o[5] = (__bf16)f1.y;
    o[6] = (__bf16)f1.z; o[7] = (__bf16)f1.w;
    *(v8bf*)(dst + i) = o;
}

// ---------------------------------------------------------------------------
// C[M,N] = A[M,K] * W[N,K]^T + bias   (K = H_ = 1024, bf16 in, f32 accum)
// MODE 0: bf16 store, head-split  [B,NH,S,HD]   (Q, K)
// MODE 1: bf16 store, transposed  [B,NH,HD,S]   (V)
// MODE 2: f32  store, row-major   [M,N]         (final output)
// Block 256 thr (8 waves), 64x64 tile, K sliced by 32, TDM double-buffered.
// ---------------------------------------------------------------------------
template <int MODE>
__global__ void GemmWmmaKernel(const __bf16* __restrict__ A,
                               const __bf16* __restrict__ W,
                               const float* __restrict__ bias,
                               void* __restrict__ dst) {
    __shared__ __align__(16) __bf16 lA[2][64 * 32];
    __shared__ __align__(16) __bf16 lB[2][64 * 32];

    const int m0   = blockIdx.x * 64;
    const int n0   = blockIdx.y * 64;
    const int tid  = threadIdx.x;
    const int wave = tid >> 5;
    const int lane = tid & 31;
    const int tm   = wave & 3;
    const int tn0  = (wave >> 2) * 2;
    const int l16  = lane & 15;
    const int half = lane >> 4;

    const __bf16* aBase = A + (size_t)m0 * H_;
    const __bf16* wBase = W + (size_t)n0 * H_;

#if !USE_TDM
    const int lrow = tid >> 2;        // 0..63
    const int lk8  = (tid & 3) * 8;   // 0,8,16,24
#endif

    // ---- stage slice 0 ----
#if USE_TDM
    if (wave == 0) {
        tdm_load_2d_bf16(lds_off_of(&lA[0][0]), aBase, 32, 64, H_);
        tdm_load_2d_bf16(lds_off_of(&lB[0][0]), wBase, 32, 64, H_);
        __builtin_amdgcn_s_wait_tensorcnt(0);
    }
#else
    *(uint4v*)(&lA[0][lrow * 32 + lk8]) = *(const uint4v*)(aBase + (size_t)lrow * H_ + lk8);
    *(uint4v*)(&lB[0][lrow * 32 + lk8]) = *(const uint4v*)(wBase + (size_t)lrow * H_ + lk8);
#endif
    __syncthreads();

    v8f acc0 = {};
    v8f acc1 = {};

    for (int k0 = 0; k0 < H_; k0 += 32) {
        const int  cur  = (k0 >> 5) & 1;
        const int  nxt  = cur ^ 1;
        const bool more = (k0 + 32) < H_;

        // ---- issue next slice DMA while computing on current ----
#if USE_TDM
        if (wave == 0 && more) {
            tdm_load_2d_bf16(lds_off_of(&lA[nxt][0]), aBase + k0 + 32, 32, 64, H_);
            tdm_load_2d_bf16(lds_off_of(&lB[nxt][0]), wBase + k0 + 32, 32, 64, H_);
        }
#else
        if (more) {
            *(uint4v*)(&lA[nxt][lrow * 32 + lk8]) =
                *(const uint4v*)(aBase + (size_t)lrow * H_ + k0 + 32 + lk8);
            *(uint4v*)(&lB[nxt][lrow * 32 + lk8]) =
                *(const uint4v*)(wBase + (size_t)lrow * H_ + k0 + 32 + lk8);
        }
#endif

        v16bf af  = load_frag16(&lA[cur][(tm * 16 + l16) * 32], half);
        v16bf bf0 = load_frag16(&lB[cur][((tn0 + 0) * 16 + l16) * 32], half);
        v16bf bf1 = load_frag16(&lB[cur][((tn0 + 1) * 16 + l16) * 32], half);

        acc0 = __builtin_amdgcn_wmma_f32_16x16x32_bf16(
                   false, af, false, bf0, (short)0, acc0, false, false);
        acc1 = __builtin_amdgcn_wmma_f32_16x16x32_bf16(
                   false, af, false, bf1, (short)0, acc1, false, false);

#if USE_TDM
        if (wave == 0) __builtin_amdgcn_s_wait_tensorcnt(0);
#endif
        __syncthreads();
    }

    // C layout: lane = column n within tile, VGPR v = row m (halves of 8 rows)
    const int rbase = m0 + tm * 16 + half * 8;
#pragma unroll
    for (int j = 0; j < 2; ++j) {
        const int n  = n0 + (tn0 + j) * 16 + l16;
        const float bv = bias[n];
        v8f acc = j ? acc1 : acc0;
#pragma unroll
        for (int v = 0; v < 8; ++v) {
            const int m = rbase + v;
            const float val = acc[v] + bv;
            if (MODE == 2) {
                ((float*)dst)[(size_t)m * H_ + n] = val;
            } else {
                const int b = m >> 11;        // / S_
                const int s = m & (S_ - 1);
                const int h = n >> 6;         // / HD_
                const int d = n & (HD_ - 1);
                if (MODE == 0) {
                    ((__bf16*)dst)[(((size_t)(b * NH_ + h) * S_) + s) * HD_ + d] =
                        (__bf16)val;
                } else {  // MODE 1: [B,NH,HD,S]
                    ((__bf16*)dst)[(((size_t)(b * NH_ + h) * HD_) + d) * S_ + s] =
                        (__bf16)val;
                }
            }
        }
    }
}

// ---------------------------------------------------------------------------
// Flash attention: grid = (S_/128, B_*NH_), 256 thr (8 waves).
// The whole block shares one key/value stream: K/V tiles (32 keys) are staged
// into LDS once per step (TDM double-buffered); each wave owns a 16-row query
// tile with online softmax. Q,K: [B,NH,S,HD] bf16; V: [B,NH,HD,S] bf16.
// ---------------------------------------------------------------------------
__global__ void FlashAttnKernel(const __bf16* __restrict__ qg,
                                const __bf16* __restrict__ kg,
                                const __bf16* __restrict__ vtg,
                                const int* __restrict__ mask,
                                __bf16* __restrict__ ctx) {
    __shared__ __align__(16) __bf16 lK[2][32 * 64];   // [key][d]
    __shared__ __align__(16) __bf16 lV[2][64 * 32];   // [d][key]
    __shared__ __align__(16) __bf16 lP[8][16 * 32];   // per-wave P tile

    const int tid   = threadIdx.x;
    const int wave  = tid >> 5;
    const int lane  = tid & 31;
    const int l16   = lane & 15;
    const int half  = lane >> 4;
    const int rhalf = half * 8;

    const int bh = blockIdx.y;           // 0..63
    const int b  = bh >> 4;              // / NH_
    const int h  = bh & (NH_ - 1);
    const int q0 = (blockIdx.x * 8 + wave) * 16;

    const __bf16* qbase = qg + ((size_t)bh * S_ + q0) * HD_;
    const __bf16* kbase = kg + (size_t)bh * S_ * HD_;
    const __bf16* vbase = vtg + (size_t)bh * HD_ * S_;
    const int* maskb    = mask + (size_t)b * S_ * S_ + (size_t)q0 * S_;

    __bf16* lPw = &lP[wave][0];

    // Q fragments (A layout): row = l16, d-chunks split by lane half
    const __bf16* qrow = qbase + (size_t)l16 * HD_;
    const v16bf a0 = load_frag16(qrow, half);        // d = 0..31
    const v16bf a1 = load_frag16(qrow + 32, half);   // d = 32..63

    // ---- stage key tile 0 ----
#if USE_TDM
    if (wave == 0) {
        tdm_load_2d_bf16(lds_off_of(&lK[0][0]), kbase, 64, 32, HD_);
        tdm_load_2d_bf16(lds_off_of(&lV[0][0]), vbase, 32, 64, S_);
        __builtin_amdgcn_s_wait_tensorcnt(0);
    }
#else
    {
        const int kr = tid >> 3, kc = (tid & 7) * 8;   // 32 rows x 64 d
        *(uint4v*)(&lK[0][kr * 64 + kc]) = *(const uint4v*)(kbase + (size_t)kr * HD_ + kc);
        const int vd = tid >> 2, vk = (tid & 3) * 8;   // 64 d x 32 keys
        *(uint4v*)(&lV[0][vd * 32 + vk]) = *(const uint4v*)(vbase + (size_t)vd * S_ + vk);
    }
#endif
    __syncthreads();

    float m_i[8], l_i[8];
#pragma unroll
    for (int v = 0; v < 8; ++v) { m_i[v] = -1e30f; l_i[v] = 0.0f; }
    v8f o0 = {}, o1 = {}, o2 = {}, o3 = {};

    const float scale = 0.125f;  // 1/sqrt(HD)

    for (int t = 0; t < S_ / 32; ++t) {
        const int  key0 = t * 32;
        const int  cur  = t & 1;
        const int  nxt  = cur ^ 1;
        const bool more = (t + 1) < (S_ / 32);

        // ---- issue next K/V tile while computing on current ----
#if USE_TDM
        if (wave == 0 && more) {
            tdm_load_2d_bf16(lds_off_of(&lK[nxt][0]), kbase + (size_t)(key0 + 32) * HD_, 64, 32, HD_);
            tdm_load_2d_bf16(lds_off_of(&lV[nxt][0]), vbase + key0 + 32, 32, 64, S_);
        }
#else
        if (more) {
            const int kr = tid >> 3, kc = (tid & 7) * 8;
            *(uint4v*)(&lK[nxt][kr * 64 + kc]) =
                *(const uint4v*)(kbase + (size_t)(key0 + 32 + kr) * HD_ + kc);
            const int vd = tid >> 2, vk = (tid & 3) * 8;
            *(uint4v*)(&lV[nxt][vd * 32 + vk]) =
                *(const uint4v*)(vbase + (size_t)vd * S_ + key0 + 32 + vk);
        }
#endif

        // ---- scores: two 16x16 tiles covering keys key0..key0+31 ----
        const __bf16* kc0 = &lK[cur][(0 * 16 + l16) * 64];
        const __bf16* kc1 = &lK[cur][(1 * 16 + l16) * 64];
        v8f s0 = {}, s1 = {};
        s0 = __builtin_amdgcn_wmma_f32_16x16x32_bf16(
                 false, a0, false, load_frag16(kc0, half), (short)0, s0, false, false);
        s0 = __builtin_amdgcn_wmma_f32_16x16x32_bf16(
                 false, a1, false, load_frag16(kc0 + 32, half), (short)0, s0, false, false);
        s1 = __builtin_amdgcn_wmma_f32_16x16x32_bf16(
                 false, a0, false, load_frag16(kc1, half), (short)0, s1, false, false);
        s1 = __builtin_amdgcn_wmma_f32_16x16x32_bf16(
                 false, a1, false, load_frag16(kc1 + 32, half), (short)0, s1, false, false);

        // ---- scale + mask + online softmax (DPP16 row reductions) ----
#pragma unroll
        for (int v = 0; v < 8; ++v) {
            const int row = rhalf + v;
            const int mA = maskb[(size_t)row * S_ + key0 + l16];
            const int mB = maskb[(size_t)row * S_ + key0 + 16 + l16];
            float sa = (mA == 0) ? -1e9f : s0[v] * scale;
            float sb = (mB == 0) ? -1e9f : s1[v] * scale;

            const float tmax = row16_max(fmaxf(sa, sb));
            const float newm = fmaxf(m_i[v], tmax);
            const float corr = __expf(m_i[v] - newm);
            const float pa = __expf(sa - newm);
            const float pb = __expf(sb - newm);
            const float rsum = row16_sum(pa + pb);

            l_i[v] = l_i[v] * corr + rsum;
            m_i[v] = newm;
            o0[v] *= corr; o1[v] *= corr; o2[v] *= corr; o3[v] *= corr;

            lPw[row * 32 + l16]      = (__bf16)pa;
            lPw[row * 32 + 16 + l16] = (__bf16)pb;
        }

        // cross-lane LDS dependence (written by column-lanes, read by row-lanes)
        asm volatile("s_wait_dscnt 0" ::: "memory");

        // ---- O += P(16x32) @ V(32x64), 4 column tiles of d ----
        const v16bf pfrag = load_frag16(lPw + l16 * 32, half);
#pragma unroll
        for (int j = 0; j < 4; ++j) {
            const v16bf vb = load_frag16(&lV[cur][(j * 16 + l16) * 32], half);
            v8f* op = (j == 0) ? &o0 : (j == 1) ? &o1 : (j == 2) ? &o2 : &o3;
            *op = __builtin_amdgcn_wmma_f32_16x16x32_bf16(
                      false, pfrag, false, vb, (short)0, *op, false, false);
        }

#if USE_TDM
        if (wave == 0) __builtin_amdgcn_s_wait_tensorcnt(0);
#endif
        __syncthreads();
    }

    // ---- normalize + store ctx[b, s, h*HD + d] (bf16) ----
#pragma unroll
    for (int j = 0; j < 4; ++j) {
        const v8f oc = (j == 0) ? o0 : (j == 1) ? o1 : (j == 2) ? o2 : o3;
        const int d = j * 16 + l16;
#pragma unroll
        for (int v = 0; v < 8; ++v) {
            const int row = rhalf + v;
            const float inv = 1.0f / l_i[v];
            ctx[((size_t)(b * S_) + q0 + row) * H_ + h * HD_ + d] =
                (__bf16)(oc[v] * inv);
        }
    }
}

// ---------------------------------------------------------------------------
// Host launcher
// ---------------------------------------------------------------------------
extern "C" void kernel_launch(void* const* d_in, const int* in_sizes, int n_in,
                              void* d_out, int out_size, void* d_ws, size_t ws_size,
                              hipStream_t stream) {
    const float* x    = (const float*)d_in[0];
    const int*   mask = (const int*)d_in[1];
    const float* Wq   = (const float*)d_in[2];
    const float* bq   = (const float*)d_in[3];
    const float* Wk   = (const float*)d_in[4];
    const float* bk   = (const float*)d_in[5];
    const float* Wv   = (const float*)d_in[6];
    const float* bv   = (const float*)d_in[7];
    const float* Wo   = (const float*)d_in[8];
    const float* bo   = (const float*)d_in[9];
    float* out = (float*)d_out;

    char* ws = (char*)d_ws;
    __bf16* xb  = (__bf16*)ws; ws += (size_t)M_ * H_ * 2;   // 16 MB
    __bf16* wqb = (__bf16*)ws; ws += (size_t)H_ * H_ * 2;   //  2 MB
    __bf16* wkb = (__bf16*)ws; ws += (size_t)H_ * H_ * 2;
    __bf16* wvb = (__bf16*)ws; ws += (size_t)H_ * H_ * 2;
    __bf16* wob = (__bf16*)ws; ws += (size_t)H_ * H_ * 2;
    __bf16* qb  = (__bf16*)ws; ws += (size_t)M_ * H_ * 2;   // [B,NH,S,HD]
    __bf16* kb  = (__bf16*)ws; ws += (size_t)M_ * H_ * 2;   // [B,NH,S,HD]
    __bf16* vtb = (__bf16*)ws; ws += (size_t)M_ * H_ * 2;   // [B,NH,HD,S]
    __bf16* ctx = (__bf16*)ws; ws += (size_t)M_ * H_ * 2;   // [B,S,H]

    // fp32 -> bf16
    CvtBf16Kernel<<<(M_ * H_) / (8 * 256), 256, 0, stream>>>(x, xb, M_ * H_);
    CvtBf16Kernel<<<(H_ * H_) / (8 * 256), 256, 0, stream>>>(Wq, wqb, H_ * H_);
    CvtBf16Kernel<<<(H_ * H_) / (8 * 256), 256, 0, stream>>>(Wk, wkb, H_ * H_);
    CvtBf16Kernel<<<(H_ * H_) / (8 * 256), 256, 0, stream>>>(Wv, wvb, H_ * H_);
    CvtBf16Kernel<<<(H_ * H_) / (8 * 256), 256, 0, stream>>>(Wo, wob, H_ * H_);

    // QKV projections
    dim3 gg(M_ / 64, H_ / 64);
    GemmWmmaKernel<0><<<gg, 256, 0, stream>>>(xb, wqb, bq, (void*)qb);
    GemmWmmaKernel<0><<<gg, 256, 0, stream>>>(xb, wkb, bk, (void*)kb);
    GemmWmmaKernel<1><<<gg, 256, 0, stream>>>(xb, wvb, bv, (void*)vtb);

    // attention
    dim3 ga(S_ / 128, B_ * NH_);
    FlashAttnKernel<<<ga, 256, 0, stream>>>(qb, kb, vtb, mask, ctx);

    // output projection (fp32 out)
    GemmWmmaKernel<2><<<gg, 256, 0, stream>>>(ctx, wob, bo, (void*)out);
}
